// RelationalTAGConv_1451698946530
// MI455X (gfx1250) — matrix operand order repaired and checked by reading, not compiled, not run on hardware
//
#include <hip/hip_runtime.h>
#include <hip/hip_bf16.h>

#define N_NODES 100000
#define N_EDGES 1000000
#define D_IN    128
#define D_CAT   384          // (K_HOPS+1)*D_IN
#define D_OUT   64
#define N_REL   4
#define OUT_STRIDE 256       // N_REL*D_OUT
#define M_BLOCKS (N_NODES / 16)   // 6250 exact

typedef __attribute__((ext_vector_type(16))) _Float16 v16h;
typedef __attribute__((ext_vector_type(8)))  _Float16 v8h;
typedef __attribute__((ext_vector_type(8)))  float    v8f;

// ---------------- utility ----------------
__global__ __launch_bounds__(256) void zero_f32(float* p, int n) {
    int i = blockIdx.x * blockDim.x + threadIdx.x;
    if (i < n) p[i] = 0.0f;
}

// ---------------- degree / norm ----------------
__global__ __launch_bounds__(256) void deg_kernel(const int* __restrict__ dst,
                                                  const unsigned char* __restrict__ mask,
                                                  float* __restrict__ deg) {
    int e = blockIdx.x * blockDim.x + threadIdx.x;
    if (e >= N_EDGES) return;
    if (mask[e]) atomicAdd(&deg[dst[e]], 1.0f);
}

__global__ __launch_bounds__(256) void norm_kernel(float* __restrict__ deg_norm) {
    int n = blockIdx.x * blockDim.x + threadIdx.x;
    if (n >= N_NODES) return;
    float d = deg_norm[n];
    deg_norm[n] = rsqrtf(fmaxf(d, 1.0f));
}

// ---------------- feature-stack init: cols [0,128)=x, [128,384)=0 ----------------
__global__ __launch_bounds__(256) void init_fstack(const float* __restrict__ x,
                                                   float* __restrict__ fs) {
    // one thread per float4 of the 384-wide row: 96 float4 per node
    long long idx = (long long)blockIdx.x * blockDim.x + threadIdx.x;
    long long total = (long long)N_NODES * 96;
    if (idx >= total) return;
    int n = (int)(idx / 96);
    int q = (int)(idx % 96);          // float4 index within row
    float4 v;
    if (q < 32) {
        v = *(const float4*)(x + (size_t)n * D_IN + q * 4);
    } else {
        v = make_float4(0.f, 0.f, 0.f, 0.f);
    }
    *(float4*)(fs + (size_t)n * D_CAT + q * 4) = v;
}

// ---------------- edge scatter: one wave per edge, 32 lanes x float4 = 128 ----------------
__global__ __launch_bounds__(256) void scatter_hop(const int* __restrict__ src,
                                                   const int* __restrict__ dst,
                                                   const unsigned char* __restrict__ mask,
                                                   const float* __restrict__ norm,
                                                   float* __restrict__ fs,
                                                   int in_col, int out_col) {
    int wid  = (blockIdx.x * blockDim.x + threadIdx.x) >> 5;  // edge id
    int lane = threadIdx.x & 31;
    if (wid >= N_EDGES) return;
    if (!mask[wid]) return;
    int s = src[wid];
    int d = dst[wid];
    float ns = norm[s];                                        // pre-norm (src side)
    float4 v = *(const float4*)(fs + (size_t)s * D_CAT + in_col + lane * 4);
    float* o = fs + (size_t)d * D_CAT + out_col + lane * 4;
    atomicAdd(o + 0, v.x * ns);
    atomicAdd(o + 1, v.y * ns);
    atomicAdd(o + 2, v.z * ns);
    atomicAdd(o + 3, v.w * ns);
}

// ---------------- post-norm: fs[n, col..col+128) *= norm[n] ----------------
__global__ __launch_bounds__(256) void postscale(float* __restrict__ fs,
                                                 const float* __restrict__ norm,
                                                 int col) {
    long long idx = (long long)blockIdx.x * blockDim.x + threadIdx.x;
    long long total = (long long)N_NODES * 32;    // float4 granularity
    if (idx >= total) return;
    int n = (int)(idx / 32);
    int q = (int)(idx % 32);
    float s = norm[n];
    float4* p = (float4*)(fs + (size_t)n * D_CAT + col + q * 4);
    float4 v = *p;
    v.x *= s; v.y *= s; v.z *= s; v.w *= s;
    *p = v;
}

// ---------------- W transpose+convert: Wt[r][n][k] = (f16) W[r][k][n] ----------------
__global__ __launch_bounds__(256) void prep_wt(const float* __restrict__ W,
                                               _Float16* __restrict__ Wt) {
    int idx = blockIdx.x * blockDim.x + threadIdx.x;    // over N_REL*D_OUT*D_CAT
    if (idx >= N_REL * D_OUT * D_CAT) return;
    int r   = idx / (D_OUT * D_CAT);
    int rem = idx % (D_OUT * D_CAT);
    int n   = rem / D_CAT;
    int k   = rem % D_CAT;
    Wt[idx] = (_Float16)W[(size_t)r * D_CAT * D_OUT + (size_t)k * D_OUT + n];
}

// ---------------- WMMA GEMM: one wave computes a 16(M) x 64(N) block ----------------
// out[m, rel*64 + n] = relu( fs[m,:] @ W[rel] + b[rel] )
__global__ __launch_bounds__(256) void gemm_wmma(const float* __restrict__ fs,
                                                 const _Float16* __restrict__ Wt,  // [64][384] f16
                                                 const float* __restrict__ bias,   // [64]
                                                 float* __restrict__ out,
                                                 int rel) {
    int wid  = (blockIdx.x * blockDim.x + threadIdx.x) >> 5;
    int lane = threadIdx.x & 31;
    if (wid >= M_BLOCKS) return;
    int m0 = wid * 16;
    int lm = lane & 15;
    int kb_a = (lane >> 4) * 8;     // A: lanes 0-15 -> K {0..7,16..23}; 16-31 -> {8..15,24..31}
    int kb_b = (lane >> 4) * 8;     // B: lanes 0-15 -> K {0..7,16..23}; 16-31 -> {8..15,24..31}

    v8f acc0 = {}, acc1 = {}, acc2 = {}, acc3 = {};
    const float* arow = fs + (size_t)(m0 + lm) * D_CAT;

    for (int k0 = 0; k0 < D_CAT; k0 += 32) {
        // ---- assemble A fragment (f32 -> f16) ----
        float4 a0 = *(const float4*)(arow + k0 + kb_a);
        float4 a1 = *(const float4*)(arow + k0 + kb_a + 4);
        float4 a2 = *(const float4*)(arow + k0 + kb_a + 16);
        float4 a3 = *(const float4*)(arow + k0 + kb_a + 20);
        v16h a;
        a[0]  = (_Float16)a0.x; a[1]  = (_Float16)a0.y; a[2]  = (_Float16)a0.z; a[3]  = (_Float16)a0.w;
        a[4]  = (_Float16)a1.x; a[5]  = (_Float16)a1.y; a[6]  = (_Float16)a1.z; a[7]  = (_Float16)a1.w;
        a[8]  = (_Float16)a2.x; a[9]  = (_Float16)a2.y; a[10] = (_Float16)a2.z; a[11] = (_Float16)a2.w;
        a[12] = (_Float16)a3.x; a[13] = (_Float16)a3.y; a[14] = (_Float16)a3.z; a[15] = (_Float16)a3.w;

        // ---- 4 N-tiles of B, pre-transposed f16 (contiguous along K) ----
        #pragma unroll
        for (int t = 0; t < 4; ++t) {
            const _Float16* brow = Wt + (size_t)(t * 16 + lm) * D_CAT + k0 + kb_b;
            v8h b0 = *(const v8h*)(brow);
            v8h b1 = *(const v8h*)(brow + 16);
            v16h b;
            #pragma unroll
            for (int i = 0; i < 8; ++i) { b[i] = b0[i]; b[8 + i] = b1[i]; }
            v8f c = (t == 0) ? acc0 : (t == 1) ? acc1 : (t == 2) ? acc2 : acc3;
            c = __builtin_amdgcn_wmma_f32_16x16x32_f16(false, a, false, b,
                                                       (short)0, c, false, false);
            if (t == 0) acc0 = c; else if (t == 1) acc1 = c; else if (t == 2) acc2 = c; else acc3 = c;
        }
    }

    // ---- epilogue: bias + relu, C/D layout: lane<16 -> M=j ; lane>=16 -> M=8+j ; N = lane%16 ----
    int mbase = (lane >> 4) * 8;
    #pragma unroll
    for (int t = 0; t < 4; ++t) {
        int n = t * 16 + lm;
        float bv = bias[n];
        v8f c = (t == 0) ? acc0 : (t == 1) ? acc1 : (t == 2) ? acc2 : acc3;
        #pragma unroll
        for (int j = 0; j < 8; ++j) {
            int m = m0 + mbase + j;
            float v = c[j] + bv;
            out[(size_t)m * OUT_STRIDE + rel * D_OUT + n] = v > 0.0f ? v : 0.0f;
        }
    }
}

// ---------------- host orchestration ----------------
extern "C" void kernel_launch(void* const* d_in, const int* in_sizes, int n_in,
                              void* d_out, int out_size, void* d_ws, size_t ws_size,
                              hipStream_t stream) {
    const float*         x    = (const float*)d_in[0];
    const int*           src  = (const int*)d_in[1];
    const int*           dst  = (const int*)d_in[2];
    const unsigned char* mask = (const unsigned char*)d_in[3];   // jnp.bool_ -> 1 byte/elem
    const float*         W    = (const float*)d_in[4];
    const float*         b    = (const float*)d_in[5];
    float*               out  = (float*)d_out;

    // workspace layout (bytes): [norm: N*4][align][fstack: N*384*4][align][Wt: 4*64*384*2]
    char* ws = (char*)d_ws;
    size_t off_norm = 0;
    size_t off_fs   = ((size_t)N_NODES * 4 + 511) & ~(size_t)511;                 // 400384
    size_t off_wt   = (off_fs + (size_t)N_NODES * D_CAT * 4 + 511) & ~(size_t)511;
    float*    norm = (float*)(ws + off_norm);
    float*    fs   = (float*)(ws + off_fs);
    _Float16* Wt   = (_Float16*)(ws + off_wt);

    const int TPB = 256;
    dim3 blk(TPB);

    // prepare transposed f16 weights once per call
    {
        int total = N_REL * D_OUT * D_CAT;
        prep_wt<<<dim3((total + TPB - 1) / TPB), blk, 0, stream>>>(W, Wt);
    }

    long long fs_init_threads = (long long)N_NODES * 96;
    long long ps_threads      = (long long)N_NODES * 32;
    long long sc_threads      = (long long)N_EDGES * 32;

    for (int r = 0; r < N_REL; ++r) {
        const unsigned char* mr = mask + (size_t)r * N_EDGES;

        // degree -> norm (in place)
        zero_f32<<<dim3((N_NODES + TPB - 1) / TPB), blk, 0, stream>>>(norm, N_NODES);
        deg_kernel<<<dim3((N_EDGES + TPB - 1) / TPB), blk, 0, stream>>>(dst, mr, norm);
        norm_kernel<<<dim3((N_NODES + TPB - 1) / TPB), blk, 0, stream>>>(norm);

        // fstack: cols 0-127 = x, cols 128-383 zeroed (agg targets)
        init_fstack<<<dim3((unsigned)((fs_init_threads + TPB - 1) / TPB)), blk, 0, stream>>>(x, fs);

        // hop 1: agg(cols 128..255) += x[src]*norm[src] ; then *= norm[dst]
        scatter_hop<<<dim3((unsigned)((sc_threads + TPB - 1) / TPB)), blk, 0, stream>>>(
            src, dst, mr, norm, fs, 0, 128);
        postscale<<<dim3((unsigned)((ps_threads + TPB - 1) / TPB)), blk, 0, stream>>>(fs, norm, 128);

        // hop 2: agg(cols 256..383) += h1[src]*norm[src] ; then *= norm[dst]
        scatter_hop<<<dim3((unsigned)((sc_threads + TPB - 1) / TPB)), blk, 0, stream>>>(
            src, dst, mr, norm, fs, 128, 256);
        postscale<<<dim3((unsigned)((ps_threads + TPB - 1) / TPB)), blk, 0, stream>>>(fs, norm, 256);

        // WMMA GEMM: [100000 x 384] @ [384 x 64] + bias, relu, write output stripe
        long long gemm_threads = (long long)M_BLOCKS * 32;
        gemm_wmma<<<dim3((unsigned)((gemm_threads + TPB - 1) / TPB)), blk, 0, stream>>>(
            fs, Wt + (size_t)r * D_OUT * D_CAT, b + (size_t)r * D_OUT, out, r);
    }
}
